// RGCN_52819507806387
// MI455X (gfx1250) — compile-verified, matrix operand order checked by reading, not run on hardware
//
#include <hip/hip_runtime.h>
#include <hip/hip_bf16.h>
#include <hip/hip_fp16.h>

typedef __attribute__((ext_vector_type(16))) _Float16 v16h;
typedef __attribute__((ext_vector_type(8)))  float    v8f;

#define NREL   16
#define NN     16000
#define NE     32000
#define FD     256
#define HD     256
#define KTOT   (NREL * FD + FD)   // 4352: [16 relation blocks | root block]
#define NGRAPH 32
#define NCLS   8

// ---------------------------------------------------------------------------
// LDS helpers: 32-bit LDS byte address for DS/async instructions
// ---------------------------------------------------------------------------
__device__ __forceinline__ unsigned lds_addr_u32(const void* p) {
    return (unsigned)(unsigned long long)
        (__attribute__((address_space(3))) const void*)p;
}

// CDNA5 async copy global->LDS (16B), tracked with ASYNCcnt
__device__ __forceinline__ void async_load_b128(unsigned lds_byte_addr,
                                                const void* gaddr) {
    asm volatile("global_load_async_to_lds_b128 %0, %1, off"
                 :: "v"(lds_byte_addr), "v"(gaddr)
                 : "memory");
}

template <int N>
__device__ __forceinline__ void wait_asynccnt() {
    asm volatile("s_wait_asynccnt %0" :: "i"(N) : "memory");
}

// ---------------------------------------------------------------------------
// Utility kernels
// ---------------------------------------------------------------------------
__global__ void k_zero(float* __restrict__ p, long long n) {
    long long i = (long long)blockIdx.x * 256 + threadIdx.x;
    if (i < n) p[i] = 0.0f;
}

__global__ void k_f2h(const float* __restrict__ x, _Float16* __restrict__ y, long long n) {
    long long i = (long long)blockIdx.x * 256 + threadIdx.x;
    if (i < n) y[i] = (_Float16)x[i];
}

// ---------------------------------------------------------------------------
// RGCN: per-relation mean aggregation (scatter) + pack into f16 A matrix
// ---------------------------------------------------------------------------
__global__ __launch_bounds__(256) void k_rgcn_scatter(
    const float* __restrict__ feat, const int* __restrict__ ei,
    const int* __restrict__ et, float* __restrict__ agg, float* __restrict__ cnt)
{
    long long gid = (long long)blockIdx.x * 256 + threadIdx.x;
    const int perEdge = FD / 4;                       // 64 chunks of float4
    int e = (int)(gid / perEdge);
    int c = (int)(gid % perEdge) * 4;
    if (e >= NE) return;
    int s = ei[e], d = ei[NE + e], r = et[e];
    const float4 v = *(const float4*)&feat[(long long)s * FD + c];
    float* dp = &agg[((long long)d * NREL + r) * FD + c];
    atomicAdd(dp + 0, v.x); atomicAdd(dp + 1, v.y);
    atomicAdd(dp + 2, v.z); atomicAdd(dp + 3, v.w);
    if (c == 0) atomicAdd(&cnt[d * NREL + r], 1.0f);
}

__global__ void k_finalize_A(const float* __restrict__ agg, const float* __restrict__ cnt,
                             const float* __restrict__ h, _Float16* __restrict__ A)
{
    long long gid = (long long)blockIdx.x * 256 + threadIdx.x;
    long long n = gid / KTOT;
    int k = (int)(gid % KTOT);
    if (n >= NN) return;
    float v;
    if (k < NREL * FD) {
        int r = k / FD;
        float c = cnt[n * NREL + r];
        v = agg[n * (long long)(NREL * FD) + k] / fmaxf(c, 1.0f);
    } else {
        v = h[n * FD + (k - NREL * FD)];              // root block = raw features
    }
    A[gid] = (_Float16)v;
}

// pack [W (R,F,H) ; root (F,H)] -> Bt (H, KTOT) f16, TRANSPOSED:
// Bt[o][k] = (k < R*F) ? W[k][o] : root[k-R*F][o]
__global__ void k_pack_w12(const float* __restrict__ W, const float* __restrict__ root,
                           _Float16* __restrict__ Bt)
{
    long long gid = (long long)blockIdx.x * 256 + threadIdx.x;
    if (gid >= (long long)HD * KTOT) return;
    long long k = gid % KTOT;
    int o = (int)(gid / KTOT);
    float v = (k < NREL * FD) ? W[k * HD + o]
                              : root[(k - NREL * FD) * HD + o];
    Bt[gid] = (_Float16)v;
}

// pack W3 (R,H,H) -> Bt (R*H, H) f16, TRANSPOSED for GEMM K=H:
// Bt[r*H+o][k] = W3[r][k][o]  so hx[n][r*H+o] = sum_k h[n][k]*W3[r][k][o]
__global__ void k_pack_w3(const float* __restrict__ W3, _Float16* __restrict__ Bt)
{
    long long gid = (long long)blockIdx.x * 256 + threadIdx.x;
    if (gid >= (long long)NREL * HD * HD) return;
    int k = (int)(gid % HD);
    int c = (int)(gid / HD);          // c = r*H + o
    int r = c / HD, o = c % HD;
    Bt[gid] = (_Float16)W3[((long long)r * HD + k) * HD + o];
}

// ---------------------------------------------------------------------------
// WMMA GEMM: C[M,Ncols] = A[M,K] (f16) x Bt[Ncols,K] (f16, pre-transposed) + bias
// Block tile 128x128, 8 waves (4 M-groups x 2 N-groups), 8 accs/wave.
// Double-buffered LDS with CDNA5 async global->LDS staging (ASYNCcnt):
// tile i+1 is in flight while tile i feeds the WMMAs.
// Requires M%128==0, K%32==0, Ncols%128==0 (true for all call sites).
// ---------------------------------------------------------------------------
template <bool OUT_HALF>
__global__ __launch_bounds__(256) void k_gemm_wmma(
    const _Float16* __restrict__ A, const _Float16* __restrict__ Bt,
    float* __restrict__ Cf, _Float16* __restrict__ Ch,
    const float* __restrict__ bias, int M, int K, int Ncols)
{
    __shared__ _Float16 As[2][128][40];  // 32 + 8 pad -> 80B rows (16B aligned)
    __shared__ _Float16 Bs[2][128][40];  // col-major tile: [col][k], same padding

    const int tid   = threadIdx.x;
    const int lane  = tid & 31;
    const int wave  = tid >> 5;
    const int wm    = wave & 3;        // M group (32 rows each)
    const int wn    = wave >> 2;       // N group (64 cols each)
    const int mlane = lane & 15;
    const int g     = lane >> 4;       // K half-group per ISA 16-bit layout

    const int m0 = blockIdx.x * 128;
    const int n0 = blockIdx.y * 128;

    // staging coordinates for this thread (4 async b128 per wave per tile)
    const int arow = tid >> 1;               // with pass p: rows tid>>1 over 2 passes? see below
    (void)arow;

    v8f acc[2][4] = {};

    const int steps = K / 32;

    // --- stage one 128x32 A tile + 128x32 B tile into buffer `buf` ---
    auto stage = [&](int it, int buf) {
        const int k0 = it * 32;
        // A: 128 rows x 4 chunks = 512 x b128, 2 passes
#pragma unroll
        for (int p = 0; p < 2; ++p) {
            int idx = p * 256 + tid;
            int row = idx >> 2;
            int ch  = (idx & 3) * 8;
            async_load_b128(lds_addr_u32(&As[buf][row][ch]),
                            &A[(long long)(m0 + row) * K + k0 + ch]);
        }
        // B: 128 cols x 4 chunks = 512 x b128, 2 passes
#pragma unroll
        for (int p = 0; p < 2; ++p) {
            int idx = p * 256 + tid;
            int col = idx >> 2;
            int ch  = (idx & 3) * 8;
            async_load_b128(lds_addr_u32(&Bs[buf][col][ch]),
                            &Bt[(long long)(n0 + col) * K + k0 + ch]);
        }
    };

    stage(0, 0);

    for (int it = 0; it < steps; ++it) {
        const int cur = it & 1;
        if (it + 1 < steps) {
            // speculative prefetch two tiles ahead (global_prefetch_b8)
            if (it + 2 < steps) {
                __builtin_prefetch(&A[(long long)(m0 + (tid >> 1)) * K + (it + 2) * 32], 0, 1);
                __builtin_prefetch(&Bt[(long long)(n0 + (tid >> 1)) * K + (it + 2) * 32], 0, 1);
            }
            stage(it + 1, cur ^ 1);       // overlap next tile with this compute
            wait_asynccnt<4>();           // tile `it` (older 4 instrs) has landed
        } else {
            wait_asynccnt<0>();
        }
        __syncthreads();                  // all waves: tile visible, prev reads done

        // A fragments: lane holds row mlane, halves K = g*8+[0..7], 16+g*8+[0..7]
        v16h afrag[2];
#pragma unroll
        for (int mi = 0; mi < 2; ++mi) {
            int r = wm * 32 + mi * 16 + mlane;
            union { v16h v; uint4 q[2]; } ua;
            ua.q[0] = *(const uint4*)&As[cur][r][g * 8];
            ua.q[1] = *(const uint4*)&As[cur][r][16 + g * 8];
            afrag[mi] = ua.v;
        }
        // B fragments: lane holds column mlane, same K striping (col-major tile)
#pragma unroll
        for (int ni = 0; ni < 4; ++ni) {
            int c = wn * 64 + ni * 16 + mlane;
            union { v16h v; uint4 q[2]; } ub;
            ub.q[0] = *(const uint4*)&Bs[cur][c][g * 8];
            ub.q[1] = *(const uint4*)&Bs[cur][c][16 + g * 8];
#pragma unroll
            for (int mi = 0; mi < 2; ++mi) {
                acc[mi][ni] = __builtin_amdgcn_wmma_f32_16x16x32_f16(
                    false, afrag[mi], false, ub.v, (short)0, acc[mi][ni],
                    false, false);
            }
        }
        __syncthreads();                  // done reading buf `cur` before reuse
    }

    // store per ISA C/D layout: VGPR v -> row v + 8*g, lane&15 -> col
#pragma unroll
    for (int mi = 0; mi < 2; ++mi) {
#pragma unroll
        for (int ni = 0; ni < 4; ++ni) {
            int colg = n0 + wn * 64 + ni * 16 + mlane;
            float badd = bias ? bias[colg] : 0.0f;
#pragma unroll
            for (int v = 0; v < 8; ++v) {
                int rowg = m0 + wm * 32 + mi * 16 + v + g * 8;
                float val = acc[mi][ni][v] + badd;
                if (OUT_HALF) Ch[(long long)rowg * Ncols + colg] = (_Float16)val;
                else          Cf[(long long)rowg * Ncols + colg] = val;
            }
        }
    }
}

// ---------------------------------------------------------------------------
// BatchNorm (population stats) + ReLU, one block per column
// ---------------------------------------------------------------------------
__global__ __launch_bounds__(256) void k_bn_relu(
    const float* __restrict__ x, const float* __restrict__ gam,
    const float* __restrict__ bet, float* __restrict__ y)
{
    __shared__ float ss[256], sq[256];
    int col = blockIdx.x, tid = threadIdx.x;
    float s = 0.0f, q = 0.0f;
    for (int n = tid; n < NN; n += 256) {
        float v = x[(long long)n * HD + col];
        s += v; q += v * v;
    }
    ss[tid] = s; sq[tid] = q;
    __syncthreads();
    for (int o = 128; o > 0; o >>= 1) {
        if (tid < o) { ss[tid] += ss[tid + o]; sq[tid] += sq[tid + o]; }
        __syncthreads();
    }
    float mu  = ss[0] / (float)NN;
    float var = sq[0] / (float)NN - mu * mu;
    float inv = rsqrtf(var + 1e-5f) * gam[col];
    float bb  = bet[col];
    for (int n = tid; n < NN; n += 256) {
        float v = (x[(long long)n * HD + col] - mu) * inv + bb;
        y[(long long)n * HD + col] = v > 0.0f ? v : 0.0f;
    }
}

// ---------------------------------------------------------------------------
// RGAT attention
// ---------------------------------------------------------------------------
__device__ inline void atomicMaxFloat(float* addr, float val) {
    int* ia = (int*)addr;
    int old = __float_as_int(*addr);
    while (__int_as_float(old) < val) {
        int assumed = old;
        old = atomicCAS(ia, assumed, __float_as_int(val));
        if (old == assumed) break;
    }
}

__global__ void k_rgat_init(float* __restrict__ m, float* __restrict__ denom) {
    int i = blockIdx.x * 256 + threadIdx.x;
    if (i < NN) { m[i] = -3.0e38f; denom[i] = 0.0f; }
}

__global__ __launch_bounds__(256) void k_rgat_alpha(
    const _Float16* __restrict__ hx, const int* __restrict__ ei,
    const int* __restrict__ et, const float* __restrict__ q,
    const float* __restrict__ kk, float* __restrict__ alpha, float* __restrict__ m)
{
    int lane = threadIdx.x & 31;
    int e = (blockIdx.x * 256 + threadIdx.x) >> 5;   // one wave per edge
    if (e >= NE) return;
    int s = ei[e], d = ei[NE + e], r = et[e];
    const _Float16* hs = hx + (long long)s * (NREL * HD) + r * HD;
    const _Float16* hd = hx + (long long)d * (NREL * HD) + r * HD;
    float acc = 0.0f;
    for (int i = lane; i < HD; i += 32)
        acc += (float)hd[i] * q[i] + (float)hs[i] * kk[i];
    for (int off = 16; off > 0; off >>= 1)
        acc += __shfl_xor(acc, off, 32);
    if (lane == 0) {
        float a = acc > 0.0f ? acc : 0.2f * acc;     // leaky_relu(., 0.2)
        alpha[e] = a;
        atomicMaxFloat(&m[d], a);
    }
}

__global__ void k_rgat_exp(const float* __restrict__ alpha, const float* __restrict__ m,
                           const int* __restrict__ ei, float* __restrict__ ev,
                           float* __restrict__ denom)
{
    int e = blockIdx.x * 256 + threadIdx.x;
    if (e >= NE) return;
    int d = ei[NE + e];
    float mu = m[d];
    if (mu < -1.0e37f) mu = 0.0f;                    // non-finite -> 0
    float x = expf(alpha[e] - mu);
    ev[e] = x;
    atomicAdd(&denom[d], x);
}

__global__ void k_bias_init(float* __restrict__ y, const float* __restrict__ b) {
    long long i = (long long)blockIdx.x * 256 + threadIdx.x;
    if (i < (long long)NN * HD) y[i] = b[(int)(i % HD)];
}

__global__ __launch_bounds__(256) void k_att_scatter(
    const _Float16* __restrict__ hx, const int* __restrict__ ei,
    const int* __restrict__ et, const float* __restrict__ ev,
    const float* __restrict__ denom, float* __restrict__ out)
{
    long long gid = (long long)blockIdx.x * 256 + threadIdx.x;
    const int perEdge = HD / 4;
    int e = (int)(gid / perEdge);
    int c = (int)(gid % perEdge) * 4;
    if (e >= NE) return;
    int s = ei[e], d = ei[NE + e], r = et[e];
    float att = ev[e] / fmaxf(denom[d], 1e-16f);
    const _Float16* hp = &hx[(long long)s * (NREL * HD) + r * HD + c];
    float* op = &out[(long long)d * HD + c];
    atomicAdd(op + 0, att * (float)hp[0]);
    atomicAdd(op + 1, att * (float)hp[1]);
    atomicAdd(op + 2, att * (float)hp[2]);
    atomicAdd(op + 3, att * (float)hp[3]);
}

// ---------------------------------------------------------------------------
// Global mean pool + final linear
// ---------------------------------------------------------------------------
__global__ void k_pool_zero(float* __restrict__ pooled, float* __restrict__ pcnt) {
    int i = blockIdx.x * 256 + threadIdx.x;
    if (i < NGRAPH * HD) pooled[i] = 0.0f;
    if (i < NGRAPH) pcnt[i] = 0.0f;
}

__global__ __launch_bounds__(256) void k_pool_scatter(
    const float* __restrict__ h, const int* __restrict__ batch,
    float* __restrict__ pooled, float* __restrict__ pcnt)
{
    long long gid = (long long)blockIdx.x * 256 + threadIdx.x;
    const int perNode = HD / 4;
    int n = (int)(gid / perNode);
    int c = (int)(gid % perNode) * 4;
    if (n >= NN) return;
    int b = batch[n];
    const float4 v = *(const float4*)&h[(long long)n * HD + c];
    float* p = &pooled[b * HD + c];
    atomicAdd(p + 0, v.x); atomicAdd(p + 1, v.y);
    atomicAdd(p + 2, v.z); atomicAdd(p + 3, v.w);
    if (c == 0) atomicAdd(&pcnt[b], 1.0f);
}

__global__ void k_final(const float* __restrict__ pooled, const float* __restrict__ pcnt,
                        const float* __restrict__ lw, const float* __restrict__ lb,
                        float* __restrict__ out)
{
    int t = threadIdx.x;
    if (t >= NGRAPH * NCLS) return;
    int g = t / NCLS, c = t % NCLS;
    float inv = 1.0f / fmaxf(pcnt[g], 1.0f);
    float s = 0.0f;
    for (int f = 0; f < HD; ++f)
        s += pooled[g * HD + f] * lw[f * NCLS + c];
    out[t] = s * inv + lb[c];
}

// ---------------------------------------------------------------------------
// Host launcher
// ---------------------------------------------------------------------------
static inline unsigned nblk(long long n) { return (unsigned)((n + 255) / 256); }

extern "C" void kernel_launch(void* const* d_in, const int* in_sizes, int n_in,
                              void* d_out, int out_size, void* d_ws, size_t ws_size,
                              hipStream_t stream)
{
    const float* x     = (const float*)d_in[0];
    const int*   ei    = (const int*)d_in[1];
    const int*   et    = (const int*)d_in[2];
    const int*   batch = (const int*)d_in[3];
    const float* W1    = (const float*)d_in[4];
    const float* root1 = (const float*)d_in[5];
    const float* b1    = (const float*)d_in[6];
    const float* g1    = (const float*)d_in[7];
    const float* be1   = (const float*)d_in[8];
    const float* W2    = (const float*)d_in[9];
    const float* root2 = (const float*)d_in[10];
    const float* b2    = (const float*)d_in[11];
    const float* g2    = (const float*)d_in[12];
    const float* be2   = (const float*)d_in[13];
    const float* W3    = (const float*)d_in[14];
    const float* q3    = (const float*)d_in[15];
    const float* k3    = (const float*)d_in[16];
    const float* b3    = (const float*)d_in[17];
    const float* g3    = (const float*)d_in[18];
    const float* be3   = (const float*)d_in[19];
    const float* lw    = (const float*)d_in[20];
    const float* lb    = (const float*)d_in[21];
    float* out = (float*)d_out;

    // workspace carve-out (256B aligned regions)
    char* ws = (char*)d_ws;
    auto carve = [&](size_t bytes) -> char* {
        char* p = ws;
        ws += (bytes + 255) & ~(size_t)255;
        return p;
    };
    float*    agg  = (float*)   carve((size_t)NN * NREL * FD * 4);       // 262 MB
    float*    cnt  = (float*)   carve((size_t)NN * NREL * 4);
    _Float16* Ah   = (_Float16*)carve((size_t)NN * KTOT * 2);            // 139 MB
    _Float16* Bh   = (_Float16*)carve((size_t)KTOT * HD * 2);            // reused for W3 pack
    float*    t    = (float*)   carve((size_t)NN * HD * 4);              // GEMM out / out3
    float*    h    = (float*)   carve((size_t)NN * HD * 4);              // activations
    _Float16* hh   = (_Float16*)carve((size_t)NN * HD * 2);
    _Float16* hx   = (_Float16*)carve((size_t)NN * NREL * HD * 2);       // 131 MB
    float*    alph = (float*)   carve((size_t)NE * 4);
    float*    evb  = (float*)   carve((size_t)NE * 4);
    float*    mmax = (float*)   carve((size_t)NN * 4);
    float*    den  = (float*)   carve((size_t)NN * 4);
    float*    pool = (float*)   carve((size_t)NGRAPH * HD * 4);
    float*    pcnt = (float*)   carve((size_t)NGRAPH * 4);

    const dim3 gemmGridH(NN / 128, HD / 128);            // (125, 2)
    const dim3 gemmGridHX(NN / 128, (NREL * HD) / 128);  // (125, 32)

    // ---------------- Layer 1: RGCN(x) ----------------
    k_zero<<<nblk((long long)NN * NREL * FD), 256, 0, stream>>>(agg, (long long)NN * NREL * FD);
    k_zero<<<nblk((long long)NN * NREL), 256, 0, stream>>>(cnt, (long long)NN * NREL);
    k_rgcn_scatter<<<nblk((long long)NE * (FD / 4)), 256, 0, stream>>>(x, ei, et, agg, cnt);
    k_finalize_A<<<nblk((long long)NN * KTOT), 256, 0, stream>>>(agg, cnt, x, Ah);
    k_pack_w12<<<nblk((long long)HD * KTOT), 256, 0, stream>>>(W1, root1, Bh);
    k_gemm_wmma<false><<<gemmGridH, 256, 0, stream>>>(Ah, Bh, t, nullptr, b1, NN, KTOT, HD);
    k_bn_relu<<<HD, 256, 0, stream>>>(t, g1, be1, h);

    // ---------------- Layer 2: RGCN(h1) ----------------
    k_zero<<<nblk((long long)NN * NREL * FD), 256, 0, stream>>>(agg, (long long)NN * NREL * FD);
    k_zero<<<nblk((long long)NN * NREL), 256, 0, stream>>>(cnt, (long long)NN * NREL);
    k_rgcn_scatter<<<nblk((long long)NE * (FD / 4)), 256, 0, stream>>>(h, ei, et, agg, cnt);
    k_finalize_A<<<nblk((long long)NN * KTOT), 256, 0, stream>>>(agg, cnt, h, Ah);
    k_pack_w12<<<nblk((long long)HD * KTOT), 256, 0, stream>>>(W2, root2, Bh);
    k_gemm_wmma<false><<<gemmGridH, 256, 0, stream>>>(Ah, Bh, t, nullptr, b2, NN, KTOT, HD);
    k_bn_relu<<<HD, 256, 0, stream>>>(t, g2, be2, h);

    // ---------------- Layer 3: RGAT(h2) ----------------
    k_pack_w3<<<nblk((long long)NREL * HD * HD), 256, 0, stream>>>(W3, Bh);
    k_f2h<<<nblk((long long)NN * HD), 256, 0, stream>>>(h, hh, (long long)NN * HD);
    k_gemm_wmma<true><<<gemmGridHX, 256, 0, stream>>>(hh, Bh, nullptr, hx, nullptr,
                                                      NN, HD, NREL * HD);
    k_rgat_init<<<nblk(NN), 256, 0, stream>>>(mmax, den);
    k_rgat_alpha<<<nblk((long long)NE * 32), 256, 0, stream>>>(hx, ei, et, q3, k3, alph, mmax);
    k_rgat_exp<<<nblk(NE), 256, 0, stream>>>(alph, mmax, ei, evb, den);
    k_bias_init<<<nblk((long long)NN * HD), 256, 0, stream>>>(t, b3);
    k_att_scatter<<<nblk((long long)NE * (HD / 4)), 256, 0, stream>>>(hx, ei, et, evb, den, t);
    k_bn_relu<<<HD, 256, 0, stream>>>(t, g3, be3, h);

    // ---------------- Pool + classifier ----------------
    k_pool_zero<<<nblk(NGRAPH * HD), 256, 0, stream>>>(pool, pcnt);
    k_pool_scatter<<<nblk((long long)NN * (HD / 4)), 256, 0, stream>>>(h, batch, pool, pcnt);
    k_final<<<1, 256, 0, stream>>>(pool, pcnt, lw, lb, out);

    (void)in_sizes; (void)n_in; (void)out_size; (void)ws_size;
}
